// Generator_48541720379508
// MI455X (gfx1250) — matrix operand order repaired and checked by reading, not compiled, not run on hardware
//
#include <hip/hip_runtime.h>
#include <hip/hip_bf16.h>

// ---------------------------------------------------------------------------
// Problem constants (from reference): S=512, B=128, H=256, V=50000, PAD=0
// ---------------------------------------------------------------------------
#define S_LEN 512
#define B_SZ  128
#define H_SZ  256
#define G4H   1024          // 4*H
#define KTOT  512           // K = [x_t | h_{t-1}] = 2*H
#define NWG_DIR 16          // workgroups per direction (hidden slices of 16)

typedef __attribute__((ext_vector_type(16))) _Float16 v16h;
typedef __attribute__((ext_vector_type(8)))  _Float16 v8h;
typedef __attribute__((ext_vector_type(4)))  _Float16 v4h;
typedef __attribute__((ext_vector_type(8)))  float    v8f;

__device__ __forceinline__ float sigf(float x) { return 1.0f / (1.0f + __expf(-x)); }

// ---------------------------------------------------------------------------
// Kernel 1: embedding gather, f32 -> f16   xh[t*B+b][H]
// ---------------------------------------------------------------------------
__global__ void embed_gather(const int* __restrict__ input,
                             const float* __restrict__ emb,
                             _Float16* __restrict__ xh) {
    int idx = blockIdx.x * blockDim.x + threadIdx.x;     // one thread per 4 elems
    const int total4 = (S_LEN * B_SZ * H_SZ) / 4;
    if (idx >= total4) return;
    int pos = idx / (H_SZ / 4);                          // (t*B + b)
    int kc  = (idx - pos * (H_SZ / 4)) * 4;
    int tok = input[pos];
    const float* e = emb + (size_t)tok * H_SZ + kc;
    v4h o;
    o[0] = (_Float16)e[0]; o[1] = (_Float16)e[1];
    o[2] = (_Float16)e[2]; o[3] = (_Float16)e[3];
    *(v4h*)(xh + (size_t)pos * H_SZ + kc) = o;
}

// ---------------------------------------------------------------------------
// Kernel 2: pack weights into WMMA B-fragment lane order, f16.
// wpack[dir][ntile(64)][kt(16)][lane(32)][s(16)]
//   col = lane&15 ; k = kt*32 + (lane>>4)*8 + (s&7) + (s>>3)*16 ; n = ntile*16 + col
//   value = k<256 ? W_ih[n][k] : W_hh[n][k-256]
// ---------------------------------------------------------------------------
__global__ void pack_weights(const float* __restrict__ wih_f,
                             const float* __restrict__ whh_f,
                             const float* __restrict__ wih_b,
                             const float* __restrict__ whh_b,
                             _Float16* __restrict__ wpack) {
    int idx = blockIdx.x * blockDim.x + threadIdx.x;     // 2*64*16*32*16 = 1048576
    if (idx >= 2 * 64 * 16 * 32 * 16) return;
    int s     = idx & 15;
    int lane  = (idx >> 4) & 31;
    int kt    = (idx >> 9) & 15;
    int ntile = (idx >> 13) & 63;
    int d     = idx >> 19;
    int n = ntile * 16 + (lane & 15);
    int k = kt * 32 + ((lane >> 4) * 8) + (s & 7) + ((s >> 3) * 16);
    const float* wih = d ? wih_b : wih_f;
    const float* whh = d ? whh_b : whh_f;
    float v = (k < H_SZ) ? wih[n * H_SZ + k] : whh[n * H_SZ + (k - H_SZ)];
    wpack[idx] = (_Float16)v;
}

// ---------------------------------------------------------------------------
// Kernel 3: zero hidden double-buffers and phase barriers (deterministic per
// launch; required because d_ws is only poisoned once, never re-initialized).
// ---------------------------------------------------------------------------
__global__ void init_state(_Float16* __restrict__ hbuf, unsigned* __restrict__ bars) {
    int idx = blockIdx.x * blockDim.x + threadIdx.x;
    const int nh = 2 * 2 * B_SZ * H_SZ;                  // 131072 halves
    if (idx < nh) hbuf[idx] = (_Float16)0.0f;
    if (idx < 32) bars[idx] = 0u;                        // 2 dirs x {count, phase, pad...}
}

// ---------------------------------------------------------------------------
// Kernel 4: bidirectional LSTM scan. grid = 32 WGs (16 fwd + 16 bwd), 256 thr.
// WG (dir, n16) owns hidden cols [n16*16, n16*16+16). Wave w owns batches
// [w*16, w*16+16). Per step: gates = [x_t | h_prev] @ Wcat^T via
// v_wmma_f32_16x16x32_f16 (K=512 -> 16 k-tiles x 4 gates = 64 WMMA / wave),
// cell state kept in registers in WMMA C-layout, h double-buffered in global,
// one atomic phase barrier per direction per step.
// ---------------------------------------------------------------------------
__global__ __launch_bounds__(256)
void lstm_scan(const _Float16* __restrict__ xh,      // [S*B][H]
               const _Float16* __restrict__ wpack,   // [2][64][16][32][16]
               const float* __restrict__ bias_f,     // [1024]
               const float* __restrict__ bias_b,     // [1024]
               _Float16* __restrict__ hbuf,          // [2 dir][2 buf][B*H]
               _Float16* __restrict__ hout,          // [2 dir][S*B][H]
               unsigned* __restrict__ bars) {        // [2 dir][16]
    const int dir  = blockIdx.x >> 4;
    const int n16  = blockIdx.x & 15;
    const int tid  = threadIdx.x;
    const int wave = tid >> 5;
    const int lane = tid & 31;
    const int j0   = n16 * 16;
    const int m0   = wave * 16;
    const float* bias = dir ? bias_b : bias_f;
    unsigned* bar = bars + dir * 16;                 // [0]=count, [1]=phase

    // 64KB LDS: B-fragments for the 4 gate column-tiles of this hidden slice.
    __shared__ _Float16 Wl[4 * 16 * 32 * 16];        // [g][kt][lane][16]

    // Stage weights once (resident for the whole 512-step scan).
    {
        const _Float16* src = wpack + (size_t)dir * 64 * 16 * 32 * 16;
        for (int g = 0; g < 4; ++g) {
            const _Float16* sp = src + (size_t)(g * 16 + n16) * (16 * 32 * 16);
            _Float16* dp = Wl + g * (16 * 32 * 16);
            for (int i = tid; i < (16 * 32 * 16) / 8; i += 256)
                ((v8h*)dp)[i] = ((const v8h*)sp)[i];
        }
    }
    __syncthreads();

    const int col = j0 + (lane & 15);
    const float bI = bias[col], bF = bias[H_SZ + col];
    const float bG = bias[2 * H_SZ + col], bO = bias[3 * H_SZ + col];

    float c[8];
#pragma unroll
    for (int r = 0; r < 8; ++r) c[r] = 0.0f;

    _Float16* hb0 = hbuf + (size_t)dir * 2 * (B_SZ * H_SZ);
    _Float16* ho  = hout + (size_t)dir * (size_t)S_LEN * B_SZ * H_SZ;

    const int arow = m0 + (lane & 15);               // A-matrix row (batch index)
    const int ksel = (lane >> 4) ? 8 : 0;            // lanes 16-31 hold K+8 block
    const int mofs = (lane >> 4) << 3;               // C-layout: lanes 16-31 -> M+8

    for (int step = 0; step < S_LEN; ++step) {
        const int t = dir ? (S_LEN - 1 - step) : step;
        const _Float16* hprev = hb0 + (step & 1) * (B_SZ * H_SZ);
        _Float16*       hnext = hb0 + ((step + 1) & 1) * (B_SZ * H_SZ);

        v8f acc[4] = {};                             // i, f, g, o accumulators

        const _Float16* xrow = xh + ((size_t)t * B_SZ + arow) * H_SZ;
        const _Float16* hrow = hprev + (size_t)arow * H_SZ;

#pragma unroll
        for (int kt = 0; kt < 16; ++kt) {
            const int kb = kt * 32 + ksel;
            // A fragment: lanes 0-15 row=lane, K = kb..kb+7 and kb+16..kb+23;
            // lanes 16-31 same rows, K shifted by 8 (per CDNA5 16-bit A layout).
            const _Float16* srcp = (kt < 8) ? (xrow + kb) : (hrow + (kb - H_SZ));
            v8h alo = *(const v8h*)(srcp);
            v8h ahi = *(const v8h*)(srcp + 16);
            v16h a;
#pragma unroll
            for (int e = 0; e < 8; ++e) { a[e] = alo[e]; a[8 + e] = ahi[e]; }
#pragma unroll
            for (int g = 0; g < 4; ++g) {
                const v16h bf = *(const v16h*)(Wl + (((g * 16 + kt) * 32) + lane) * 16);
                acc[g] = __builtin_amdgcn_wmma_f32_16x16x32_f16(
                    false, a, false, bf, (short)0, acc[g], false, false);
            }
        }

        // LSTM cell update; c/h live in WMMA C-layout registers.
#pragma unroll
        for (int r = 0; r < 8; ++r) {
            float iv = sigf(acc[0][r] + bI);
            float fv = sigf(acc[1][r] + bF);
            float gv = tanhf(acc[2][r] + bG);
            float ov = sigf(acc[3][r] + bO);
            c[r] = fv * c[r] + iv * gv;
            float hv = ov * tanhf(c[r]);
            int m = m0 + r + mofs;
            _Float16 h16 = (_Float16)hv;
            hnext[(size_t)m * H_SZ + col] = h16;
            ho[((size_t)t * B_SZ + m) * H_SZ + col] = h16;
        }

        // Direction-wide phase barrier (16 WGs): publish h_t before step+1.
        __threadfence();
        __syncthreads();
        if (tid == 0) {
            unsigned arr = atomicAdd(&bar[0], 1u);
            if (arr == (unsigned)(NWG_DIR - 1)) {
                bar[0] = 0u;
                __threadfence();
                atomicAdd(&bar[1], 1u);
            }
            while (atomicAdd(&bar[1], 0u) < (unsigned)(step + 1)) {
                __builtin_amdgcn_s_sleep(1);
            }
        }
        __syncthreads();
    }
}

// ---------------------------------------------------------------------------
// Kernel 5: linear head + sigmoid + threshold against noise.
// out[0:65536] = rationale_selected_node, out[65536:131072] = rationale_selected
// ---------------------------------------------------------------------------
__global__ void head_select(const _Float16* __restrict__ hout,
                            const float* __restrict__ wlin,
                            const float* __restrict__ blin,
                            const float* __restrict__ noise,
                            float* __restrict__ out) {
    int idx = blockIdx.x * blockDim.x + threadIdx.x;     // (t*B + b)
    if (idx >= S_LEN * B_SZ) return;
    const _Float16* hf = hout + (size_t)idx * H_SZ;
    const _Float16* hb = hout + (size_t)S_LEN * B_SZ * H_SZ + (size_t)idx * H_SZ;
    float acc = blin[0];
#pragma unroll 4
    for (int j = 0; j < H_SZ; ++j)
        acc += (float)hf[j] * wlin[j] + (float)hb[j] * wlin[H_SZ + j];
    float p = sigf(acc);
    float sel = (noise[idx] < p) ? 1.0f : 0.0f;
    out[idx] = sel;                                      // [S,B,1]
    out[S_LEN * B_SZ + idx] = sel;                       // [S,B]
}

// ---------------------------------------------------------------------------
// Kernel 6: per-column stream compaction via wave32 ballot prefix-sum.
// out[131072:196608] = rationales (PAD=0), out[196608:196736] = lengths
// ---------------------------------------------------------------------------
__global__ void compact_scan(const int* __restrict__ input, float* __restrict__ out) {
    const float* sel = out + S_LEN * B_SZ;
    float* rat  = out + 2 * S_LEN * B_SZ;
    float* lens = out + 3 * S_LEN * B_SZ;
    const int b = blockIdx.x;
    const int lane = threadIdx.x;

    for (int t = lane; t < S_LEN; t += 32) rat[t * B_SZ + b] = 0.0f;   // PAD fill

    int base = 0;
    for (int c0 = 0; c0 < S_LEN; c0 += 32) {
        int t = c0 + lane;
        bool s = sel[t * B_SZ + b] > 0.5f;
        unsigned mask = (unsigned)__ballot(s);           // wave32: low 32 bits
        unsigned pre = __popc(mask & ((1u << lane) - 1u));
        if (s) rat[(base + pre) * B_SZ + b] = (float)input[t * B_SZ + b];
        base += __popc(mask);
    }
    if (lane == 0) lens[b] = (float)base;
}

// ---------------------------------------------------------------------------
// Launcher. Workspace layout (bytes):
//   [0)              xh     : S*B*H f16          = 33,554,432
//   [33,554,432)     wpack  : 2*64*16*32*16 f16  =  2,097,152
//   [35,651,584)     hbuf   : 2*2*B*H f16        =    262,144
//   [35,913,728)     hout   : 2*S*B*H f16        = 67,108,864
//   [103,022,592)    bars   : 32 u32             =        128
// ---------------------------------------------------------------------------
extern "C" void kernel_launch(void* const* d_in, const int* in_sizes, int n_in,
                              void* d_out, int out_size, void* d_ws, size_t ws_size,
                              hipStream_t stream) {
    const int*   input  = (const int*)d_in[0];
    const float* noise  = (const float*)d_in[1];
    const float* emb    = (const float*)d_in[2];
    const float* W_ih_f = (const float*)d_in[3];
    const float* W_hh_f = (const float*)d_in[4];
    const float* b_f    = (const float*)d_in[5];
    const float* W_ih_b = (const float*)d_in[6];
    const float* W_hh_b = (const float*)d_in[7];
    const float* b_b    = (const float*)d_in[8];
    const float* W_lin  = (const float*)d_in[9];
    const float* b_lin  = (const float*)d_in[10];
    float* out = (float*)d_out;

    char* ws = (char*)d_ws;
    _Float16* xh    = (_Float16*)(ws);
    _Float16* wpack = (_Float16*)(ws + 33554432);
    _Float16* hbuf  = (_Float16*)(ws + 35651584);
    _Float16* hout  = (_Float16*)(ws + 35913728);
    unsigned* bars  = (unsigned*)(ws + 103022592);

    const int n4 = (S_LEN * B_SZ * H_SZ) / 4;
    embed_gather<<<(n4 + 255) / 256, 256, 0, stream>>>(input, emb, xh);
    pack_weights<<<(2 * 64 * 16 * 32 * 16 + 255) / 256, 256, 0, stream>>>(
        W_ih_f, W_hh_f, W_ih_b, W_hh_b, wpack);
    init_state<<<(2 * 2 * B_SZ * H_SZ + 255) / 256, 256, 0, stream>>>(hbuf, bars);
    lstm_scan<<<2 * NWG_DIR, 256, 0, stream>>>(xh, wpack, b_f, b_b, hbuf, hout, bars);
    head_select<<<(S_LEN * B_SZ + 255) / 256, 256, 0, stream>>>(hout, W_lin, b_lin, noise, out);
    compact_scan<<<B_SZ, 32, 0, stream>>>(input, out);
}
